// Set2Set_50878182588793
// MI455X (gfx1250) — matrix operand order repaired and checked by reading, not compiled, not run on hardware
//
#include <hip/hip_runtime.h>
#include <hip/hip_bf16.h>
#include <math.h>

#define G      512
#define D      256
#define KTOT   768     // 2D (q_star) + D (h)
#define GATES  1024    // 4D
#define NSTEPS 3

typedef __attribute__((ext_vector_type(16))) _Float16 v16h;
typedef __attribute__((ext_vector_type(8)))  float    v8f;

// ---- monotonic float<->uint mapping for integer atomicMax on floats ----
__device__ __forceinline__ unsigned ordF(float f) {
    unsigned u = __float_as_uint(f);
    return (u & 0x80000000u) ? ~u : (u | 0x80000000u);
}
__device__ __forceinline__ float unordF(unsigned u) {
    return (u & 0x80000000u) ? __uint_as_float(u & 0x7FFFFFFFu)
                             : __uint_as_float(~u);
}
__device__ __forceinline__ float sigm(float x) { return 1.0f / (1.0f + expf(-x)); }

// ---------------------------------------------------------------------------
// Pack W = [W_ih | W_hh] (f32, row-major [1024 x 768]) into f16 tiles laid out
// exactly as the WMMA B operand (32x16, K-major per lane):
//   tile t = n_tile*24 + k_tile, 512 halves per tile, index = t*512 + lane*16 + j
//   N = n_tile*16 + (lane&15),  K = k_tile*32 + 16*(lane>=16) + j
// => each lane reads its 16 halves as one contiguous 32B run (2x b128).
// ---------------------------------------------------------------------------
__global__ void pack_weights(const float* __restrict__ Wih,
                             const float* __restrict__ Whh,
                             _Float16* __restrict__ Wp) {
    int tid = blockIdx.x * blockDim.x + threadIdx.x;
    if (tid >= GATES * KTOT) return;
    int t    = tid >> 9;          // tile (512 halves each)
    int r    = tid & 511;
    int lane = r >> 4;
    int j    = r & 15;
    int n_tile = t / 24;
    int k_tile = t % 24;
    int n = n_tile * 16 + (lane & 15);
    int k = k_tile * 32 + ((lane >= 16) ? 16 : 0) + j;
    float v = (k < 512) ? Wih[n * 512 + k] : Whh[n * 256 + (k - 512)];
    Wp[tid] = (_Float16)v;
}

__global__ void pack_bias(const float* __restrict__ bih,
                          const float* __restrict__ bhh,
                          float* __restrict__ bias) {
    int i = blockIdx.x * blockDim.x + threadIdx.x;
    if (i < GATES) bias[i] = bih[i] + bhh[i];
}

__global__ void zero_f(float* __restrict__ p, int n) {
    int i = blockIdx.x * blockDim.x + threadIdx.x;
    if (i < n) p[i] = 0.0f;
}

__global__ void step_init(float* __restrict__ rraw,
                          float* __restrict__ esum,
                          unsigned* __restrict__ emax) {
    int i = blockIdx.x * blockDim.x + threadIdx.x;
    if (i < G * D) rraw[i] = 0.0f;
    if (i < G) { esum[i] = 0.0f; emax[i] = 0u; }   // 0 < ordF(-inf)
}

// ---------------------------------------------------------------------------
// LSTM cell: gates = [q_star | h_in] @ W^T + bias, then elementwise update.
// One wave (32 lanes) per 16 graphs x 16 hidden dims; computes the i/f/g/o
// tiles for the same (m,d) block so the cell update fuses in-register.
// grid = (G/16, D/16), block = 32.  EXEC is all-ones (no divergence) as
// required by WMMA.
// ---------------------------------------------------------------------------
__global__ void lstm_wmma(const float* __restrict__ qstar,   // [G, 2D] (prev step)
                          const float* __restrict__ hin,     // [G, D]  (prev step)
                          float* __restrict__ hout,          // [G, D]  (new h)
                          float* __restrict__ cbuf,          // [G, D]  in-place
                          const _Float16* __restrict__ Wp,   // packed B tiles
                          const float* __restrict__ bias) {  // [4D]
    const int lane = threadIdx.x;             // 0..31
    const int m0   = blockIdx.x * 16;         // graph tile base
    const int dt   = blockIdx.y;              // hidden-dim tile index (0..15)
    const int d0   = dt * 16;

    v8f acc0 = {}, acc1 = {}, acc2 = {}, acc3 = {};

    const int m   = m0 + (lane & 15);         // A row for this lane
    const int khi = (lane >> 4) << 4;         // +16 for upper half-wave

    for (int kt = 0; kt < 24; ++kt) {
        // ---- A operand: X[m, k] with X = [q_star | h_in], f32 -> f16 ----
        // lane layout (matches packed B): j=0..15 -> K=kb+j
        const int kb  = kt * 32 + khi;
        const int kb1 = kb + 8;
        const float* p0 = (kb  < 512) ? (qstar + (size_t)m * 512 + kb)
                                      : (hin   + (size_t)m * 256 + (kb  - 512));
        const float* p1 = (kb1 < 512) ? (qstar + (size_t)m * 512 + kb1)
                                      : (hin   + (size_t)m * 256 + (kb1 - 512));
        float4 q0 = *(const float4*)(p0);
        float4 q1 = *(const float4*)(p0 + 4);
        float4 q2 = *(const float4*)(p1);
        float4 q3 = *(const float4*)(p1 + 4);
        v16h A;
        A[0]=(_Float16)q0.x; A[1]=(_Float16)q0.y; A[2]=(_Float16)q0.z; A[3]=(_Float16)q0.w;
        A[4]=(_Float16)q1.x; A[5]=(_Float16)q1.y; A[6]=(_Float16)q1.z; A[7]=(_Float16)q1.w;
        A[8]=(_Float16)q2.x; A[9]=(_Float16)q2.y; A[10]=(_Float16)q2.z; A[11]=(_Float16)q2.w;
        A[12]=(_Float16)q3.x; A[13]=(_Float16)q3.y; A[14]=(_Float16)q3.z; A[15]=(_Float16)q3.w;

        // ---- B operand: pre-swizzled tiles, one per gate (n_tile = g*16+dt) ----
        const size_t lo = (size_t)lane * 16;
        const v16h B0 = *(const v16h*)(Wp + ((size_t)(( 0 + dt) * 24 + kt)) * 512 + lo);
        const v16h B1 = *(const v16h*)(Wp + ((size_t)((16 + dt) * 24 + kt)) * 512 + lo);
        const v16h B2 = *(const v16h*)(Wp + ((size_t)((32 + dt) * 24 + kt)) * 512 + lo);
        const v16h B3 = *(const v16h*)(Wp + ((size_t)((48 + dt) * 24 + kt)) * 512 + lo);

        acc0 = __builtin_amdgcn_wmma_f32_16x16x32_f16(false, A, false, B0, (short)0, acc0, false, false);
        acc1 = __builtin_amdgcn_wmma_f32_16x16x32_f16(false, A, false, B1, (short)0, acc1, false, false);
        acc2 = __builtin_amdgcn_wmma_f32_16x16x32_f16(false, A, false, B2, (short)0, acc2, false, false);
        acc3 = __builtin_amdgcn_wmma_f32_16x16x32_f16(false, A, false, B3, (short)0, acc3, false, false);
    }

    // ---- fused LSTM cell epilogue ----
    // C/D layout: vgpr r -> M = r + 8*(lane>=16), N = lane&15
    union { v8f v; float f[8]; } ai, af, ag, ao;
    ai.v = acc0; af.v = acc1; ag.v = acc2; ao.v = acc3;
    const int n  = d0 + (lane & 15);
    const float bi = bias[0 * 256 + n];
    const float bf = bias[1 * 256 + n];
    const float bg = bias[2 * 256 + n];
    const float bo = bias[3 * 256 + n];
    const int mrow0 = m0 + ((lane >> 4) << 3);
    for (int r = 0; r < 8; ++r) {
        const int idx = (mrow0 + r) * 256 + n;
        float iv = sigm(ai.f[r] + bi);
        float fv = sigm(af.f[r] + bf);
        float gv = tanhf(ag.f[r] + bg);
        float ov = sigm(ao.f[r] + bo);
        float cN = fv * cbuf[idx] + iv * gv;
        cbuf[idx] = cN;
        hout[idx] = ov * tanhf(cN);
    }
}

// ---------------------------------------------------------------------------
// e[n] = <features[n,:], h[batch[n],:]>  (wave per node, float4 streaming),
// plus per-graph running max via integer atomicMax on the ordered mapping.
// ---------------------------------------------------------------------------
__global__ void attn_dot(const float* __restrict__ feat,
                         const int*   __restrict__ bidx,
                         const float* __restrict__ hq,
                         float*       __restrict__ evec,
                         unsigned*    __restrict__ emax,
                         int N) {
    int gid  = blockIdx.x * blockDim.x + threadIdx.x;
    int node = gid >> 5;
    int lane = threadIdx.x & 31;
    if (node >= N) return;
    int gph = bidx[node];
    const float4* f4 = (const float4*)(feat + (size_t)node * D);
    const float4* h4 = (const float4*)(hq + (size_t)gph * D);
    if (node + 8 < N) __builtin_prefetch(feat + (size_t)(node + 8) * D, 0, 1);
    float4 a = f4[lane],      b = h4[lane];
    float  s = a.x*b.x + a.y*b.y + a.z*b.z + a.w*b.w;
    a = f4[lane + 32];        b = h4[lane + 32];
    s += a.x*b.x + a.y*b.y + a.z*b.z + a.w*b.w;
    for (int off = 16; off > 0; off >>= 1) s += __shfl_xor(s, off, 32);
    if (lane == 0) {
        evec[node] = s;
        atomicMax(emax + gph, ordF(s));
    }
}

// e[i] := exp(e[i] - max[g]);  esum[g] += e[i]
__global__ void attn_expsum(float* __restrict__ evec,
                            const unsigned* __restrict__ emax,
                            const int* __restrict__ bidx,
                            float* __restrict__ esum,
                            int N) {
    int i = blockIdx.x * blockDim.x + threadIdx.x;
    if (i >= N) return;
    int g = bidx[i];
    float wv = expf(evec[i] - unordF(emax[g]));
    evec[i] = wv;
    atomicAdd(esum + g, wv);
}

// r_raw[g,:] += w[n] * features[n,:]   (wave per node, float atomics)
__global__ void attn_weighted(const float* __restrict__ feat,
                              const float* __restrict__ evec,
                              const int*   __restrict__ bidx,
                              float*       __restrict__ rraw,
                              int N) {
    int gid  = blockIdx.x * blockDim.x + threadIdx.x;
    int node = gid >> 5;
    int lane = threadIdx.x & 31;
    if (node >= N) return;
    int g = bidx[node];
    float wv = evec[node];
    const float4* f4 = (const float4*)(feat + (size_t)node * D);
    float* rb = rraw + (size_t)g * D;
    float4 a = f4[lane];
    atomicAdd(rb + 4*lane + 0, wv * a.x);
    atomicAdd(rb + 4*lane + 1, wv * a.y);
    atomicAdd(rb + 4*lane + 2, wv * a.z);
    atomicAdd(rb + 4*lane + 3, wv * a.w);
    a = f4[lane + 32];
    atomicAdd(rb + 4*(lane + 32) + 0, wv * a.x);
    atomicAdd(rb + 4*(lane + 32) + 1, wv * a.y);
    atomicAdd(rb + 4*(lane + 32) + 2, wv * a.z);
    atomicAdd(rb + 4*(lane + 32) + 3, wv * a.w);
}

// q_star = [ h | r / (sum + 1e-16) ]
__global__ void finalize_step(const float* __restrict__ hq,
                              const float* __restrict__ rraw,
                              const float* __restrict__ esum,
                              float* __restrict__ qstar) {
    int i = blockIdx.x * blockDim.x + threadIdx.x;
    if (i >= G * D) return;
    int g = i >> 8, d = i & 255;
    qstar[g * 512 + d]       = hq[i];
    qstar[g * 512 + 256 + d] = rraw[i] / (esum[g] + 1e-16f);
}

extern "C" void kernel_launch(void* const* d_in, const int* in_sizes, int n_in,
                              void* d_out, int out_size, void* d_ws, size_t ws_size,
                              hipStream_t stream) {
    const float* feat = (const float*)d_in[0];
    const int*   bidx = (const int*)  d_in[1];
    const float* Wih  = (const float*)d_in[2];
    const float* Whh  = (const float*)d_in[3];
    const float* bih  = (const float*)d_in[4];
    const float* bhh  = (const float*)d_in[5];
    float* qstar = (float*)d_out;          // [G, 2D] — q_star IS the output
    const int N = in_sizes[1];             // batch_index length

    // workspace sub-allocation (256B aligned)
    char* w = (char*)d_ws;
    auto sub = [&](size_t bytes) -> void* {
        void* p = (void*)w;
        w += (bytes + 255) & ~(size_t)255;
        return p;
    };
    _Float16* Wp   = (_Float16*)sub((size_t)GATES * KTOT * sizeof(_Float16));
    float*    bias = (float*)   sub(GATES * sizeof(float));
    float*    hb0  = (float*)   sub((size_t)G * D * sizeof(float));
    float*    hb1  = (float*)   sub((size_t)G * D * sizeof(float));
    float*    cb   = (float*)   sub((size_t)G * D * sizeof(float));
    float*    evec = (float*)   sub((size_t)N * sizeof(float));
    unsigned* emax = (unsigned*)sub(G * sizeof(unsigned));
    float*    esum = (float*)   sub(G * sizeof(float));
    float*    rraw = (float*)   sub((size_t)G * D * sizeof(float));

    // one-time prep (re-done every launch: deterministic)
    pack_weights<<<(GATES * KTOT + 255) / 256, 256, 0, stream>>>(Wih, Whh, Wp);
    pack_bias   <<<(GATES + 255) / 256,        256, 0, stream>>>(bih, bhh, bias);
    zero_f<<<(G * D + 255) / 256,     256, 0, stream>>>(hb0, G * D);
    zero_f<<<(G * D + 255) / 256,     256, 0, stream>>>(cb, G * D);
    zero_f<<<(G * 2 * D + 255) / 256, 256, 0, stream>>>(qstar, G * 2 * D);

    const int waveThreads = N * 32;          // one wave32 per node
    for (int s = 0; s < NSTEPS; ++s) {
        float* hin  = (s & 1) ? hb1 : hb0;
        float* hout = (s & 1) ? hb0 : hb1;
        step_init<<<(G * D + 255) / 256, 256, 0, stream>>>(rraw, esum, emax);
        lstm_wmma<<<dim3(G / 16, D / 16), 32, 0, stream>>>(qstar, hin, hout, cb, Wp, bias);
        attn_dot     <<<(waveThreads + 255) / 256, 256, 0, stream>>>(feat, bidx, hout, evec, emax, N);
        attn_expsum  <<<(N + 255) / 256,           256, 0, stream>>>(evec, emax, bidx, esum, N);
        attn_weighted<<<(waveThreads + 255) / 256, 256, 0, stream>>>(feat, evec, bidx, rraw, N);
        finalize_step<<<(G * D + 255) / 256,       256, 0, stream>>>(hout, rraw, esum, qstar);
    }
}
